// GumbelDecoderEncoder_75634374082633
// MI455X (gfx1250) — matrix operand order repaired and checked by reading, not compiled, not run on hardware
//
#include <hip/hip_runtime.h>

typedef __attribute__((ext_vector_type(16))) __bf16 bf16x16;
typedef __attribute__((ext_vector_type(8)))  __bf16 bf16x8;
typedef __attribute__((ext_vector_type(8)))  float  f32x8;

#define EDIM 256
#define VDIM 128
#define TSTEPS 16
#define GDIM 768          // 3*EDIM gate width
#define MROWS 32          // batch rows per workgroup (2 WMMA M-tiles)
#define HS 260            // f32 LDS row stride (h, gin, ghn)
#define BS 264            // bf16 LDS row stride (hb, embb); 264*2B = 33*16B
#define SRZS 516          // f32 LDS row stride for fused r|z sums (512 cols)
#define LGS 132           // f32 LDS row stride for logits

#define WMMA_BF16(a,b,c) \
  __builtin_amdgcn_wmma_f32_16x16x32_bf16(false,(a),false,(b),(short)0,(c),false,false)

__device__ __forceinline__ bf16x16 ld_afrag(const __bf16* rowp, int koff) {
  // A-frag k order used here, same as B-frag (16x32 bf16): lane half g holds K = 16g..16g+15 (koff%32 == 8g)
  bf16x8 lo = *(const bf16x8*)(rowp + koff + (koff & 31));
  bf16x8 hi = *(const bf16x8*)(rowp + koff + (koff & 31) + 8);
  return __builtin_shufflevector(lo, hi, 0,1,2,3,4,5,6,7,8,9,10,11,12,13,14,15);
}

__device__ __forceinline__ float sigmoidf_(float v) { return 1.0f / (1.0f + __expf(-v)); }

// Fused gi/gh GEMMs for one GRU step over MROWS=32 rows.
// embb,hb: LDS bf16 [32][BS].  Wih,Whh: global bf16 [768][256] row-major (== W, used as B=W^T).
// Writes: srz[32][SRZS] = (gi+gh+bih+bhh) for r|z cols 0..511; gin/ghn[32][HS] = n-parts + bias.
__device__ __forceinline__ void gru_gemms(
    const __bf16* __restrict__ embb, const __bf16* __restrict__ hb,
    const __bf16* __restrict__ Wih,  const __bf16* __restrict__ Whh,
    const float*  __restrict__ bih,  const float*  __restrict__ bhh,
    float* __restrict__ srz, float* __restrict__ gin, float* __restrict__ ghn, int tid)
{
  const int lane = tid & 31;
  const int wave = tid >> 5;
  const int g = lane >> 4;     // lane half
  const int m = lane & 15;
  for (int nt = wave; nt < 48; nt += 8) {           // disjoint N-tiles per wave
    f32x8 ai0 = {}, ai1 = {}, ah0 = {}, ah1 = {};
    const __bf16* wih_p = Wih + (nt*16 + m)*EDIM + g*16;
    const __bf16* whh_p = Whh + (nt*16 + m)*EDIM + g*16;
#pragma unroll
    for (int kt = 0; kt < 8; ++kt) {
      bf16x16 bfi = *(const bf16x16*)(wih_p + kt*32);   // B-frag: W[n][32kt+16g .. +15]
      bf16x16 bfh = *(const bf16x16*)(whh_p + kt*32);
      bf16x16 ae0 = ld_afrag(embb + (0*16 + m)*BS, kt*32 + 8*g);
      bf16x16 ae1 = ld_afrag(embb + (1*16 + m)*BS, kt*32 + 8*g);
      bf16x16 ahf0 = ld_afrag(hb + (0*16 + m)*BS, kt*32 + 8*g);
      bf16x16 ahf1 = ld_afrag(hb + (1*16 + m)*BS, kt*32 + 8*g);
      ai0 = WMMA_BF16(ae0, bfi, ai0);
      ai1 = WMMA_BF16(ae1, bfi, ai1);
      ah0 = WMMA_BF16(ahf0, bfh, ah0);
      ah1 = WMMA_BF16(ahf1, bfh, ah1);
    }
    const int c = nt*16 + m;
    if (nt < 32) {                                    // r and z regions: store fused sum
      const float bsum = bih[c] + bhh[c];
#pragma unroll
      for (int v = 0; v < 8; ++v) {
        srz[(     v + 8*g)*SRZS + c] = ai0[v] + ah0[v] + bsum;
        srz[(16 + v + 8*g)*SRZS + c] = ai1[v] + ah1[v] + bsum;
      }
    } else {                                          // n region: keep gi_n / gh_n separate
      const int cn = c - 512;
      const float b1 = bih[c], b2 = bhh[c];
#pragma unroll
      for (int v = 0; v < 8; ++v) {
        gin[(     v + 8*g)*HS + cn] = ai0[v] + b1;
        gin[(16 + v + 8*g)*HS + cn] = ai1[v] + b1;
        ghn[(     v + 8*g)*HS + cn] = ah0[v] + b2;
        ghn[(16 + v + 8*g)*HS + cn] = ah1[v] + b2;
      }
    }
  }
}

// logits = h @ We2d^T + be2d   ([32,256] x [256,128]); We2d global bf16 [128][256]
__device__ __forceinline__ void logits_gemm(
    const __bf16* __restrict__ hb, const __bf16* __restrict__ We2d,
    const float* __restrict__ be2d, float* __restrict__ logits, int tid)
{
  const int lane = tid & 31;
  const int nt = tid >> 5;     // 8 waves -> 8 N-tiles
  const int g = lane >> 4;
  const int m = lane & 15;
  f32x8 a0 = {}, a1 = {};
  const __bf16* wp = We2d + (nt*16 + m)*EDIM + g*16;
#pragma unroll
  for (int kt = 0; kt < 8; ++kt) {
    bf16x16 bf = *(const bf16x16*)(wp + kt*32);
    bf16x16 h0 = ld_afrag(hb + (0*16 + m)*BS, kt*32 + 8*g);
    bf16x16 h1 = ld_afrag(hb + (1*16 + m)*BS, kt*32 + 8*g);
    a0 = WMMA_BF16(h0, bf, a0);
    a1 = WMMA_BF16(h1, bf, a1);
  }
  const int c = nt*16 + m;
  const float bb = be2d[c];
#pragma unroll
  for (int v = 0; v < 8; ++v) {
    logits[(     v + 8*g)*LGS + c] = a0[v] + bb;
    logits[(16 + v + 8*g)*LGS + c] = a1[v] + bb;
  }
}

__global__ __launch_bounds__(256) void gumbel_dec_enc_fused(
    const float* __restrict__ x, const float* __restrict__ noise,
    const __bf16* __restrict__ dWih, const __bf16* __restrict__ dWhh,
    const __bf16* __restrict__ dWe2d, const __bf16* __restrict__ dWd2eT,
    const __bf16* __restrict__ eWih, const __bf16* __restrict__ eWhh,
    const __bf16* __restrict__ eWd2eT,
    const float* __restrict__ d_bd2e, const float* __restrict__ d_bih,
    const float* __restrict__ d_bhh,  const float* __restrict__ d_be2d,
    const float* __restrict__ e_bd2e, const float* __restrict__ e_bih,
    const float* __restrict__ e_bhh,
    float* __restrict__ out, int Btot)
{
  __shared__ float  h[MROWS*HS];
  __shared__ float  ostate[MROWS*EDIM];
  __shared__ float  srz[MROWS*SRZS];
  __shared__ float  gin[MROWS*HS];
  __shared__ float  ghn[MROWS*HS];
  __shared__ float  logits[MROWS*LGS];
  __shared__ alignas(32) __bf16 hb[MROWS*BS];    // 32B-aligned base: A-frags use ds b128 loads
  __shared__ alignas(32) __bf16 embb[MROWS*BS];
  __shared__ float  redv[MROWS*8];
  __shared__ int    redi[MROWS*8];
  __shared__ int    tok[MROWS];
  __shared__ int    alive[MROWS];
  __shared__ int    Nv[MROWS];
  __shared__ int    tokh[TSTEPS][MROWS];

  const int tid = threadIdx.x;
  const int r0  = blockIdx.x * MROWS;
  const int row = tid >> 3;     // 0..31
  const int s8  = tid & 7;

  // ---- init: h = x, decoder bookkeeping ----
#pragma unroll
  for (int i = 0; i < 32; ++i) {
    int c = s8*32 + i;
    float xv = x[(size_t)(r0 + row)*EDIM + c];
    h[row*HS + c]  = xv;
    hb[row*BS + c] = (__bf16)xv;
    ostate[row*EDIM + c] = 0.0f;
  }
  if (tid < MROWS) { tok[tid] = -1; alive[tid] = 1; Nv[tid] = TSTEPS; }
  __syncthreads();

  // ================= decoder =================
  for (int t = 0; t < TSTEPS; ++t) {
    // emb = one-hot gather of Wd2e^T row + bias (zero token -> bias only)
    {
      const int tk = tok[row];
#pragma unroll
      for (int i = 0; i < 32; ++i) {
        int c = s8*32 + i;
        float ev = d_bd2e[c];
        if (tk >= 0) ev += (float)dWd2eT[tk*EDIM + c];
        embb[row*BS + c] = (__bf16)ev;
      }
    }
    __syncthreads();
    gru_gemms(embb, hb, dWih, dWhh, d_bih, d_bhh, srz, gin, ghn, tid);
    __syncthreads();
    // GRU gates
#pragma unroll
    for (int i = 0; i < 32; ++i) {
      int c = s8*32 + i;
      float r  = sigmoidf_(srz[row*SRZS + c]);
      float z  = sigmoidf_(srz[row*SRZS + 256 + c]);
      float nn = tanhf(gin[row*HS + c] + r * ghn[row*HS + c]);
      float hv = (1.0f - z)*nn + z*h[row*HS + c];
      h[row*HS + c]  = hv;
      hb[row*BS + c] = (__bf16)hv;
    }
    __syncthreads();
    logits_gemm(hb, dWe2d, d_be2d, logits, tid);
    __syncthreads();
    // gumbel + argmax partials (softmax/TAU is argmax-invariant)
    {
      float best = -3.4e38f; int bi = 0;
#pragma unroll
      for (int i = 0; i < 16; ++i) {
        int v = s8*16 + i;
        float u  = noise[((size_t)t*Btot + (r0 + row))*VDIM + v];
        float gv = -__logf(-__logf(u + 1e-10f) + 1e-10f);
        float val = logits[row*LGS + v] + gv;
        if (val > best) { best = val; bi = v; }
      }
      redv[row*8 + s8] = best;
      redi[row*8 + s8] = bi;
    }
    __syncthreads();
    if (tid < MROWS) {
      float best = redv[tid*8]; int bi = redi[tid*8];
#pragma unroll
      for (int q = 1; q < 8; ++q) {
        float vq = redv[tid*8 + q];
        if (vq > best) { best = vq; bi = redi[tid*8 + q]; }
      }
      if (bi == 0) { if (t < Nv[tid]) Nv[tid] = t; alive[tid] = 0; }
      int ut = alive[tid] ? bi : -1;   // dead rows feed zero token
      tok[tid] = ut;
      tokh[t][tid] = ut;
    }
    __syncthreads();
    // write utterance one-hot [B,T,V]
    {
      const int ut = tok[row];
#pragma unroll
      for (int i = 0; i < 16; ++i) {
        int v = s8*16 + i;
        out[((size_t)(r0 + row)*TSTEPS + t)*VDIM + v] = (v == ut) ? 1.0f : 0.0f;
      }
    }
    __syncthreads();
  }

  // ================= encoder =================
#pragma unroll
  for (int i = 0; i < 32; ++i) {
    int c = s8*32 + i;
    h[row*HS + c]  = 0.0f;
    hb[row*BS + c] = (__bf16)0.0f;
  }
  if (tid < MROWS) alive[tid] = 1;
  __syncthreads();

  for (int t = 0; t < TSTEPS; ++t) {
    {
      const int tk = tokh[t][row];
#pragma unroll
      for (int i = 0; i < 32; ++i) {
        int c = s8*32 + i;
        float ev = e_bd2e[c];
        if (tk >= 0) ev += (float)eWd2eT[tk*EDIM + c];
        embb[row*BS + c] = (__bf16)ev;
      }
    }
    __syncthreads();
    gru_gemms(embb, hb, eWih, eWhh, e_bih, e_bhh, srz, gin, ghn, tid);
    __syncthreads();
    {
      const int al = alive[row];            // alive BEFORE this step's sieve update
#pragma unroll
      for (int i = 0; i < 32; ++i) {
        int c = s8*32 + i;
        float r  = sigmoidf_(srz[row*SRZS + c]);
        float z  = sigmoidf_(srz[row*SRZS + 256 + c]);
        float nn = tanhf(gin[row*HS + c] + r * ghn[row*HS + c]);
        float hv = (1.0f - z)*nn + z*h[row*HS + c];
        h[row*HS + c]  = hv;
        hb[row*BS + c] = (__bf16)hv;
        if (al) ostate[row*EDIM + c] = hv;  // out_state = where(alive, state, out_state)
      }
    }
    __syncthreads();
    if (tid < MROWS) {
      if (tokh[t][tid] == 0) alive[tid] = 0;   // u_t[:,0] > 0.5
    }
    __syncthreads();
  }

  // ---- outputs: N (int32 bits) and reconst ----
  const size_t uttN = (size_t)Btot * TSTEPS * VDIM;
  if (tid < MROWS) ((int*)out)[uttN + r0 + tid] = Nv[tid];
  const size_t roff = uttN + (size_t)Btot;
#pragma unroll
  for (int i = 0; i < 32; ++i) {
    int c = s8*32 + i;
    out[roff + (size_t)(r0 + row)*EDIM + c] = ostate[row*EDIM + c];
  }
}

// ---- one-time weight conversion into workspace ----
__global__ void cvt_bf16(const float* __restrict__ src, __bf16* __restrict__ dst, int n) {
  int i = blockIdx.x * blockDim.x + threadIdx.x;
  if (i < n) dst[i] = (__bf16)src[i];
}
__global__ void cvt_bf16_T(const float* __restrict__ src, __bf16* __restrict__ dst) {
  // src [EDIM][VDIM] -> dst [VDIM][EDIM]
  int i = blockIdx.x * blockDim.x + threadIdx.x;
  if (i < VDIM*EDIM) {
    int v = i / EDIM, e = i % EDIM;
    dst[i] = (__bf16)src[e*VDIM + v];
  }
}

extern "C" void kernel_launch(void* const* d_in, const int* in_sizes, int n_in,
                              void* d_out, int out_size, void* d_ws, size_t ws_size,
                              hipStream_t stream) {
  const float* x        = (const float*)d_in[0];
  /* d_in[1] global_idxes: unused by reference */
  const float* noise    = (const float*)d_in[2];
  const float* dec_Wd2e = (const float*)d_in[3];
  const float* dec_bd2e = (const float*)d_in[4];
  const float* dec_Wih  = (const float*)d_in[5];
  const float* dec_Whh  = (const float*)d_in[6];
  const float* dec_bih  = (const float*)d_in[7];
  const float* dec_bhh  = (const float*)d_in[8];
  const float* dec_We2d = (const float*)d_in[9];
  const float* dec_be2d = (const float*)d_in[10];
  const float* enc_Wd2e = (const float*)d_in[11];
  const float* enc_bd2e = (const float*)d_in[12];
  const float* enc_Wih  = (const float*)d_in[13];
  const float* enc_Whh  = (const float*)d_in[14];
  const float* enc_bih  = (const float*)d_in[15];
  const float* enc_bhh  = (const float*)d_in[16];
  const int Btot = in_sizes[0] / EDIM;

  __bf16* ws      = (__bf16*)d_ws;
  __bf16* dWih_b  = ws;
  __bf16* dWhh_b  = dWih_b  + GDIM*EDIM;
  __bf16* dWe2d_b = dWhh_b  + GDIM*EDIM;
  __bf16* dWd2eT  = dWe2d_b + VDIM*EDIM;
  __bf16* eWih_b  = dWd2eT  + VDIM*EDIM;
  __bf16* eWhh_b  = eWih_b  + GDIM*EDIM;
  __bf16* eWd2eT  = eWhh_b  + GDIM*EDIM;

  const int cb = 256;
  cvt_bf16  <<<(GDIM*EDIM + cb - 1)/cb, cb, 0, stream>>>(dec_Wih, dWih_b, GDIM*EDIM);
  cvt_bf16  <<<(GDIM*EDIM + cb - 1)/cb, cb, 0, stream>>>(dec_Whh, dWhh_b, GDIM*EDIM);
  cvt_bf16  <<<(VDIM*EDIM + cb - 1)/cb, cb, 0, stream>>>(dec_We2d, dWe2d_b, VDIM*EDIM);
  cvt_bf16_T<<<(VDIM*EDIM + cb - 1)/cb, cb, 0, stream>>>(dec_Wd2e, dWd2eT);
  cvt_bf16  <<<(GDIM*EDIM + cb - 1)/cb, cb, 0, stream>>>(enc_Wih, eWih_b, GDIM*EDIM);
  cvt_bf16  <<<(GDIM*EDIM + cb - 1)/cb, cb, 0, stream>>>(enc_Whh, eWhh_b, GDIM*EDIM);
  cvt_bf16_T<<<(VDIM*EDIM + cb - 1)/cb, cb, 0, stream>>>(enc_Wd2e, eWd2eT);

  gumbel_dec_enc_fused<<<Btot / MROWS, 256, 0, stream>>>(
      x, noise, dWih_b, dWhh_b, dWe2d_b, dWd2eT, eWih_b, eWhh_b, eWd2eT,
      dec_bd2e, dec_bih, dec_bhh, dec_be2d, enc_bd2e, enc_bih, enc_bhh,
      (float*)d_out, Btot);

  (void)n_in; (void)out_size; (void)ws_size;
}
